// GAT_72499047956828
// MI455X (gfx1250) — compile-verified, matrix operand order checked by reading, not run on hardware
//
#include <hip/hip_runtime.h>
#include <hip/hip_bf16.h>

// ---------------------------------------------------------------------------
// GAT 2-layer forward for MI455X (gfx1250).
//  - GEMMs via v_wmma_f32_16x16x32_bf16 (wave32, 16x16 tiles, f32 accum).
//    Full-width N block tile: x is read exactly once from HBM; weights stay
//    L2-resident. K loop fully unrolled -> WMMA-dense inner body.
//  - Edge softmax/aggregation via L2-resident f32 atomics (192MB L2 holds
//    the [N,256] accumulator), coalesced wave-per-edge gathers.
//  - att vectors folded into W so x@W_dst is never materialized.
// ---------------------------------------------------------------------------

typedef __attribute__((ext_vector_type(16))) __bf16 v16bf;
typedef __attribute__((ext_vector_type(8)))  float  v8f;

union FragBF16 { uint4 q[2]; v16bf bf; };

__device__ __forceinline__ unsigned short f2bf(float f) {
    unsigned u = __float_as_uint(f);
    unsigned r = u + 0x7FFFu + ((u >> 16) & 1u);   // round-to-nearest-even
    return (unsigned short)(r >> 16);
}

// order-preserving float<->uint map for atomicMax-based segment max
__device__ __forceinline__ unsigned f2ord(float f) {
    unsigned b = __float_as_uint(f);
    return (b & 0x80000000u) ? ~b : (b | 0x80000000u);
}
__device__ __forceinline__ float ord2f(unsigned o) {
    unsigned b = (o & 0x80000000u) ? (o & 0x7FFFFFFFu) : ~o;
    return __uint_as_float(b);
}

__device__ __forceinline__ float lrelu(float v) { return v > 0.f ? v : 0.2f * v; }

// ---------------------------------------------------------------------------
// Utility fills / conversions
// ---------------------------------------------------------------------------
__global__ void k_fill_u32(unsigned* __restrict__ p, unsigned val, size_t n) {
    size_t i = (size_t)blockIdx.x * blockDim.x + threadIdx.x;
    if (i < n) p[i] = val;
}

__global__ void k_cvt_bf16(const float* __restrict__ src, unsigned short* __restrict__ dst, size_t n) {
    size_t i = (size_t)blockIdx.x * blockDim.x + threadIdx.x;
    if (i < n) dst[i] = f2bf(src[i]);
}

// vec[k*H+h] = sum_c W[k, h*C+c] * att[h*C+c]    (fold att into weights)
__global__ void k_fold_att(const float* __restrict__ W, const float* __restrict__ att,
                           float* __restrict__ vec, int K, int H, int C) {
    int t = blockIdx.x * blockDim.x + threadIdx.x;
    if (t >= K * H) return;
    int k = t / H, h = t % H;
    const float* wr = W + (size_t)k * H * C + (size_t)h * C;
    const float* ar = att + (size_t)h * C;
    float s = 0.f;
    for (int c = 0; c < C; ++c) s += wr[c] * ar[c];
    vec[(size_t)k * H + h] = s;
}

// ---------------------------------------------------------------------------
// Per-node attention scalars: a_s[n,h] = x[n,:] @ vs[:,h]; a_d likewise.
// One wave per node; lane-strided (coalesced) K loop, shuffle reduction.
// ---------------------------------------------------------------------------
template <int H>
__global__ __launch_bounds__(256) void k_att_scores(
    const float* __restrict__ X, const float* __restrict__ vs, const float* __restrict__ vd,
    float* __restrict__ as_, float* __restrict__ ad_, int Nn, int K) {
    int wave = (int)((blockIdx.x * (unsigned)blockDim.x + threadIdx.x) >> 5);
    int lane = threadIdx.x & 31;
    if (wave >= Nn) return;
    float accS[H], accD[H];
#pragma unroll
    for (int h = 0; h < H; ++h) { accS[h] = 0.f; accD[h] = 0.f; }
    const float* xr = X + (size_t)wave * K;
    for (int k = lane; k < K; k += 32) {
        float xv = xr[k];
        const float* vsr = vs + (size_t)k * H;
        const float* vdr = vd + (size_t)k * H;
#pragma unroll
        for (int h = 0; h < H; ++h) {
            accS[h] += xv * vsr[h];
            accD[h] += xv * vdr[h];
        }
    }
#pragma unroll
    for (int off = 16; off > 0; off >>= 1) {
#pragma unroll
        for (int h = 0; h < H; ++h) {
            accS[h] += __shfl_xor(accS[h], off);
            accD[h] += __shfl_xor(accD[h], off);
        }
    }
    if (lane == 0) {
#pragma unroll
        for (int h = 0; h < H; ++h) {
            as_[(size_t)wave * H + h] = accS[h];
            ad_[(size_t)wave * H + h] = accD[h];
        }
    }
}

// ---------------------------------------------------------------------------
// WMMA GEMM: Y[M,NO] = bf16(A[M,K=256]) @ Bw[K,NO]   (Bw already bf16)
// Block: 256 threads = 8 waves; block tile 64(M) x NO(N) (full width -> A
// read exactly once). Wave (wm 0..3, wn 0..1) owns NO/32 16x16 accumulators.
// A and transposed-B tiles staged in LDS so each fragment half-lane loads
// exactly two contiguous 16B chunks (ISA 7.12.2: lanes 0-15 -> K{0..7,16..23},
// lanes 16-31 -> K{8..15,24..31}). K loop fully unrolled (8 steps of 32).
// ---------------------------------------------------------------------------
template <int NO>
__global__ __launch_bounds__(256) void k_gemm_bf16_wmma(
    const float* __restrict__ A, const unsigned short* __restrict__ Bw,
    float* __restrict__ Y, int M) {
    constexpr int K  = 256;
    constexpr int NT = NO / 32;                 // 16-col subtiles per wave
    __shared__ __attribute__((aligned(16))) unsigned short As[64][40];  // [m][k]
    __shared__ __attribute__((aligned(16))) unsigned short Bs[NO][40];  // [n][k] (transposed)

    int tid  = threadIdx.x;
    int wave = tid >> 5, lane = tid & 31;
    int wm = wave >> 1, wn = wave & 1;
    int blockM = blockIdx.y * 64;               // gridDim.x == 1
    int l15 = lane & 15, khalf = lane >> 4;

    const v8f vzero = {0.f, 0.f, 0.f, 0.f, 0.f, 0.f, 0.f, 0.f};
    v8f acc[NT];
#pragma unroll
    for (int t = 0; t < NT; ++t) acc[t] = vzero;

    int r0 = tid >> 3;                          // 0..31 (A-stage row)
    int ca = (tid & 7) * 4;                     // 0..28 (A-stage col)
    int kb = tid >> 3;                          // 0..31 (B-stage k)
    int n0 = (tid & 7) * 8;                     // 0..56 (B-stage n)

#pragma unroll
    for (int kk = 0; kk < K; kk += 32) {
        // --- stage A tile (64 x 32 f32 -> bf16) ---
#pragma unroll
        for (int rr = r0; rr < 64; rr += 32) {
            int grow = blockM + rr;
            float4 v = make_float4(0.f, 0.f, 0.f, 0.f);
            if (grow < M) v = *(const float4*)(A + (size_t)grow * K + kk + ca);
            unsigned p0 = (unsigned)f2bf(v.x) | ((unsigned)f2bf(v.y) << 16);
            unsigned p1 = (unsigned)f2bf(v.z) | ((unsigned)f2bf(v.w) << 16);
            *(uint2*)&As[rr][ca] = make_uint2(p0, p1);
        }
        // --- prefetch next K-step of A (global_prefetch_b8) ---
        if (kk + 32 < K) {
            int grow = blockM + r0;
            if (grow < M) __builtin_prefetch(A + (size_t)grow * K + kk + 32 + ca, 0, 3);
        }
        // --- stage B tile transposed (32 x NO bf16 -> Bs[n][k]) ---
#pragma unroll
        for (int nn = 0; nn < NO; nn += 64) {
            uint4 wv = *(const uint4*)(Bw + (size_t)(kk + kb) * NO + nn + n0);
            int nb = nn + n0;
            Bs[nb + 0][kb] = (unsigned short)(wv.x);
            Bs[nb + 1][kb] = (unsigned short)(wv.x >> 16);
            Bs[nb + 2][kb] = (unsigned short)(wv.y);
            Bs[nb + 3][kb] = (unsigned short)(wv.y >> 16);
            Bs[nb + 4][kb] = (unsigned short)(wv.z);
            Bs[nb + 5][kb] = (unsigned short)(wv.z >> 16);
            Bs[nb + 6][kb] = (unsigned short)(wv.w);
            Bs[nb + 7][kb] = (unsigned short)(wv.w >> 16);
        }
        __syncthreads();

        FragBF16 fa;
        fa.q[0] = *(const uint4*)&As[wm * 16 + l15][khalf * 8];
        fa.q[1] = *(const uint4*)&As[wm * 16 + l15][16 + khalf * 8];
#pragma unroll
        for (int t = 0; t < NT; ++t) {
            int col = wn * (NO / 2) + t * 16 + l15;
            FragBF16 fb;
            fb.q[0] = *(const uint4*)&Bs[col][khalf * 8];
            fb.q[1] = *(const uint4*)&Bs[col][16 + khalf * 8];
            acc[t] = __builtin_amdgcn_wmma_f32_16x16x32_bf16(false, fa.bf, false, fb.bf,
                                                             (short)0, acc[t], false, false);
        }
        __syncthreads();
    }

    // C/D layout: VGPR r -> M = r + (lane>=16)*8 ; N = lane&15
    int lhi = lane >> 4;
#pragma unroll
    for (int t = 0; t < NT; ++t) {
#pragma unroll
        for (int r = 0; r < 8; ++r) {
            int grow = blockM + wm * 16 + lhi * 8 + r;
            if (grow < M)
                Y[(size_t)grow * NO + wn * (NO / 2) + t * 16 + l15] = acc[t][r];
        }
    }
}

// ---------------------------------------------------------------------------
// Edge pass 1: segment max of leaky_relu(a_s[src]+a_d[dst]) per (dst, head)
// ---------------------------------------------------------------------------
template <int H>
__global__ void k_edge_max(const int* __restrict__ src, const int* __restrict__ dst,
                           const float* __restrict__ as_, const float* __restrict__ ad_,
                           unsigned* __restrict__ menc, size_t EH) {
    size_t t = (size_t)blockIdx.x * blockDim.x + threadIdx.x;
    if (t >= EH) return;
    size_t e = t / H;
    int h = (int)(t % H);
    int s = src[e], d = dst[e];
    float v = lrelu(as_[(size_t)s * H + h] + ad_[(size_t)d * H + h]);
    atomicMax(&menc[(size_t)d * H + h], f2ord(v));
}

// ---------------------------------------------------------------------------
// Edge pass 2: s[dst,h] += exp(e - m[dst,h])
// ---------------------------------------------------------------------------
template <int H>
__global__ void k_edge_expsum(const int* __restrict__ src, const int* __restrict__ dst,
                              const float* __restrict__ as_, const float* __restrict__ ad_,
                              const unsigned* __restrict__ menc, float* __restrict__ ssum,
                              size_t EH) {
    size_t t = (size_t)blockIdx.x * blockDim.x + threadIdx.x;
    if (t >= EH) return;
    size_t e = t / H;
    int h = (int)(t % H);
    int s = src[e], d = dst[e];
    float v = lrelu(as_[(size_t)s * H + h] + ad_[(size_t)d * H + h]);
    float ex = __expf(v - ord2f(menc[(size_t)d * H + h]));
    atomicAdd(&ssum[(size_t)d * H + h], ex);
}

// ---------------------------------------------------------------------------
// Edge pass 3 (layer 1, H=8, C=32): one wave per edge, lane = channel.
// acc[dst, h*32+lane] += alpha * XS[src, h*32+lane]  (128B coalesced)
// ---------------------------------------------------------------------------
__global__ __launch_bounds__(256) void k_edge_agg1(
    const int* __restrict__ src, const int* __restrict__ dst,
    const float* __restrict__ as_, const float* __restrict__ ad_,
    const unsigned* __restrict__ menc, const float* __restrict__ ssum,
    const float* __restrict__ XS, float* __restrict__ acc, int E) {
    int wave = (int)((blockIdx.x * 256u + threadIdx.x) >> 5);
    int lane = threadIdx.x & 31;
    if (wave >= E) return;
    int s = src[wave], d = dst[wave];
    const float* xr = XS + (size_t)s * 256;
    float* ar = acc + (size_t)d * 256;
#pragma unroll
    for (int h = 0; h < 8; ++h) {
        float v = lrelu(as_[(size_t)s * 8 + h] + ad_[(size_t)d * 8 + h]);
        float alpha = __expf(v - ord2f(menc[(size_t)d * 8 + h])) /
                      (ssum[(size_t)d * 8 + h] + 1e-16f);
        atomicAdd(&ar[h * 32 + lane], alpha * xr[h * 32 + lane]);
    }
}

// ---------------------------------------------------------------------------
// Edge pass 3 (layer 2, H=1, C=128): one wave per edge, 4 channels/lane.
// ---------------------------------------------------------------------------
__global__ __launch_bounds__(256) void k_edge_agg2(
    const int* __restrict__ src, const int* __restrict__ dst,
    const float* __restrict__ as_, const float* __restrict__ ad_,
    const unsigned* __restrict__ menc, const float* __restrict__ ssum,
    const float* __restrict__ XS, float* __restrict__ acc, int E) {
    int wave = (int)((blockIdx.x * 256u + threadIdx.x) >> 5);
    int lane = threadIdx.x & 31;
    if (wave >= E) return;
    int s = src[wave], d = dst[wave];
    float v = lrelu(as_[s] + ad_[d]);
    float alpha = __expf(v - ord2f(menc[d])) / (ssum[d] + 1e-16f);
    float4 xv = *(const float4*)(XS + (size_t)s * 128 + lane * 4);
    float* o = acc + (size_t)d * 128 + lane * 4;
    atomicAdd(o + 0, alpha * xv.x);
    atomicAdd(o + 1, alpha * xv.y);
    atomicAdd(o + 2, alpha * xv.z);
    atomicAdd(o + 3, alpha * xv.w);
}

__global__ void k_bias_relu(float* __restrict__ acc, const float* __restrict__ b, size_t total) {
    size_t i = (size_t)blockIdx.x * blockDim.x + threadIdx.x;
    if (i < total) acc[i] = fmaxf(acc[i] + b[i & 255], 0.f);
}

__global__ void k_out_bias(float* __restrict__ out, const float* __restrict__ b, size_t total) {
    size_t i = (size_t)blockIdx.x * blockDim.x + threadIdx.x;
    if (i < total) out[i] = out[i] + b[i & 127];
}

// ---------------------------------------------------------------------------
extern "C" void kernel_launch(void* const* d_in, const int* in_sizes, int n_in,
                              void* d_out, int out_size, void* d_ws, size_t ws_size,
                              hipStream_t stream) {
    const float* x      = (const float*)d_in[0];
    const int*   ei     = (const int*)d_in[1];   // [2,E] (harness delivers int32)
    const float* W1_src = (const float*)d_in[2];
    const float* W1_dst = (const float*)d_in[3];
    const float* att1_s = (const float*)d_in[4];
    const float* att1_d = (const float*)d_in[5];
    const float* b1     = (const float*)d_in[6];
    const float* W2_src = (const float*)d_in[7];
    const float* W2_dst = (const float*)d_in[8];
    const float* att2_s = (const float*)d_in[9];
    const float* att2_d = (const float*)d_in[10];
    const float* b2     = (const float*)d_in[11];
    float* out = (float*)d_out;

    const int N = in_sizes[0] / 256;   // 100000
    const int E = in_sizes[1] / 2;     // 1600000
    const int* src = ei;
    const int* dst = ei + E;

    // ---- workspace layout ----
    auto alignUp = [](size_t v) { return (v + 255) & ~(size_t)255; };
    char* ws = (char*)d_ws;
    size_t o = 0;
    size_t XS1_off = o; o = alignUp(o + (size_t)N * 256 * 4);   // xs layer1 (reused as XS2)
    size_t ACC_off = o; o = alignUp(o + (size_t)N * 256 * 4);   // layer1 accum -> layer2 input
    size_t AS1_off = o; o = alignUp(o + (size_t)N * 8 * 4);
    size_t AD1_off = o; o = alignUp(o + (size_t)N * 8 * 4);
    size_t M1_off  = o; o = alignUp(o + (size_t)N * 8 * 4);
    size_t S1_off  = o; o = alignUp(o + (size_t)N * 8 * 4);
    size_t AS2_off = o; o = alignUp(o + (size_t)N * 4);
    size_t AD2_off = o; o = alignUp(o + (size_t)N * 4);
    size_t M2_off  = o; o = alignUp(o + (size_t)N * 4);
    size_t S2_off  = o; o = alignUp(o + (size_t)N * 4);
    size_t WC1_off = o; o = alignUp(o + (size_t)256 * 256 * 2); // bf16 W1_src
    size_t WC2_off = o; o = alignUp(o + (size_t)256 * 128 * 2); // bf16 W2_src
    size_t VS1_off = o; o = alignUp(o + (size_t)256 * 8 * 4);
    size_t VD1_off = o; o = alignUp(o + (size_t)256 * 8 * 4);
    size_t VS2_off = o; o = alignUp(o + (size_t)256 * 4);
    size_t VD2_off = o; o = alignUp(o + (size_t)256 * 4);
    (void)ws_size; (void)n_in; (void)out_size;

    float*          XS1 = (float*)(ws + XS1_off);
    float*          XS2 = (float*)(ws + XS1_off);   // alias: XS1 dead before GEMM2
    float*          ACC = (float*)(ws + ACC_off);
    float*          AS1 = (float*)(ws + AS1_off);
    float*          AD1 = (float*)(ws + AD1_off);
    unsigned*       M1  = (unsigned*)(ws + M1_off);
    float*          S1  = (float*)(ws + S1_off);
    float*          AS2 = (float*)(ws + AS2_off);
    float*          AD2 = (float*)(ws + AD2_off);
    unsigned*       M2  = (unsigned*)(ws + M2_off);
    float*          S2  = (float*)(ws + S2_off);
    unsigned short* WC1 = (unsigned short*)(ws + WC1_off);
    unsigned short* WC2 = (unsigned short*)(ws + WC2_off);
    float*          VS1 = (float*)(ws + VS1_off);
    float*          VD1 = (float*)(ws + VD1_off);
    float*          VS2 = (float*)(ws + VS2_off);
    float*          VD2 = (float*)(ws + VD2_off);

    const unsigned NEG_INF_ORD = 0x007FFFFFu;      // f2ord(-inf)
    auto blocks = [](size_t n, int t) { return (unsigned)((n + t - 1) / t); };

    // ---- init ----
    k_fill_u32<<<blocks((size_t)N * 8, 256), 256, 0, stream>>>(M1, NEG_INF_ORD, (size_t)N * 8);
    k_fill_u32<<<blocks((size_t)N * 8, 256), 256, 0, stream>>>((unsigned*)S1, 0u, (size_t)N * 8);
    k_fill_u32<<<blocks((size_t)N * 256, 256), 256, 0, stream>>>((unsigned*)ACC, 0u, (size_t)N * 256);
    k_fill_u32<<<blocks((size_t)N, 256), 256, 0, stream>>>(M2, NEG_INF_ORD, (size_t)N);
    k_fill_u32<<<blocks((size_t)N, 256), 256, 0, stream>>>((unsigned*)S2, 0u, (size_t)N);
    k_fill_u32<<<blocks((size_t)N * 128, 256), 256, 0, stream>>>((unsigned*)out, 0u, (size_t)N * 128);

    // ---- weight prep ----
    k_cvt_bf16<<<blocks(256 * 256, 256), 256, 0, stream>>>(W1_src, WC1, 256 * 256);
    k_cvt_bf16<<<blocks(256 * 128, 256), 256, 0, stream>>>(W2_src, WC2, 256 * 128);
    k_fold_att<<<blocks(256 * 8, 256), 256, 0, stream>>>(W1_src, att1_s, VS1, 256, 8, 32);
    k_fold_att<<<blocks(256 * 8, 256), 256, 0, stream>>>(W1_dst, att1_d, VD1, 256, 8, 32);
    k_fold_att<<<blocks(256, 256), 256, 0, stream>>>(W2_src, att2_s, VS2, 256, 1, 128);
    k_fold_att<<<blocks(256, 256), 256, 0, stream>>>(W2_dst, att2_d, VD2, 256, 1, 128);

    // ================= layer 1 =================
    k_att_scores<8><<<blocks((size_t)N * 32, 256), 256, 0, stream>>>(x, VS1, VD1, AS1, AD1, N, 256);
    {
        dim3 g(1, (N + 63) / 64);
        k_gemm_bf16_wmma<256><<<g, 256, 0, stream>>>(x, WC1, XS1, N);
    }
    k_edge_max<8><<<blocks((size_t)E * 8, 256), 256, 0, stream>>>(src, dst, AS1, AD1, M1, (size_t)E * 8);
    k_edge_expsum<8><<<blocks((size_t)E * 8, 256), 256, 0, stream>>>(src, dst, AS1, AD1, M1, S1, (size_t)E * 8);
    k_edge_agg1<<<blocks((size_t)E * 32, 256), 256, 0, stream>>>(src, dst, AS1, AD1, M1, S1, XS1, ACC, E);
    k_bias_relu<<<blocks((size_t)N * 256, 256), 256, 0, stream>>>(ACC, b1, (size_t)N * 256);

    // ================= layer 2 =================
    k_att_scores<1><<<blocks((size_t)N * 32, 256), 256, 0, stream>>>(ACC, VS2, VD2, AS2, AD2, N, 256);
    {
        dim3 g(1, (N + 63) / 64);
        k_gemm_bf16_wmma<128><<<g, 256, 0, stream>>>(ACC, WC2, XS2, N);
    }
    k_edge_max<1><<<blocks((size_t)E, 256), 256, 0, stream>>>(src, dst, AS2, AD2, M2, (size_t)E);
    k_edge_expsum<1><<<blocks((size_t)E, 256), 256, 0, stream>>>(src, dst, AS2, AD2, M2, S2, (size_t)E);
    k_edge_agg2<<<blocks((size_t)E * 32, 256), 256, 0, stream>>>(src, dst, AS2, AD2, M2, S2, XS2, out, E);
    k_out_bias<<<blocks((size_t)N * 128, 256), 256, 0, stream>>>(out, b2, (size_t)N * 128);
}
